// SparseOT_62466004353059
// MI455X (gfx1250) — matrix operand order repaired and checked by reading, not compiled, not run on hardware
//
#include <hip/hip_runtime.h>

#define BI   128
#define BT   128
#define RD   36     // real image regions
#define RP   48     // padded rows for 16x16 tiling
#define WD   48     // caption words
#define DD   1024   // feature dim
#define LAMB_F   0.05f
#define INV_LAMB 20.0f
#define EPS_F    1e-6f
#define NPASS    8

typedef __attribute__((ext_vector_type(16))) __bf16 v16bf;
typedef __attribute__((ext_vector_type(8)))  __bf16 bf16x8;
typedef __attribute__((ext_vector_type(8)))  float  v8f;

static __device__ __forceinline__ v16bf concat8(bf16x8 lo, bf16x8 hi) {
  return __builtin_shufflevector(lo, hi, 0, 1, 2, 3, 4, 5, 6, 7,
                                         8, 9, 10, 11, 12, 13, 14, 15);
}

static __device__ __forceinline__ v8f wmma_bf16(v16bf a, v16bf b, v8f c) {
  // D = A(16x32 bf16) * B(32x16 bf16) + C(16x16 f32)
  return __builtin_amdgcn_wmma_f32_16x16x32_bf16(false, a, false, b,
                                                 (short)0, c, false, false);
}

// ---------------------------------------------------------------------------
// Kernel 1: L2-normalize each row (image region / caption word) and split the
// fp32 result into bf16 hi + bf16 lo residual. Image rows are padded 36 -> 48
// with zeros so the GEMM kernel needs no bounds checks.
// ---------------------------------------------------------------------------
__global__ __launch_bounds__(256) void normsplit_kernel(
    const float* __restrict__ imgs, const float* __restrict__ caps,
    __bf16* __restrict__ Ahi, __bf16* __restrict__ Alo,
    __bf16* __restrict__ Bhi, __bf16* __restrict__ Blo)
{
  __shared__ float red[256];
  const int b   = blockIdx.x;
  const int tid = (int)threadIdx.x;
  const int NIMG = BI * RP;   // 6144 padded image rows

  const float* src = nullptr;
  __bf16* dh;
  __bf16* dl;
  bool valid;
  if (b < NIMG) {
    const int ii = b / RP, r = b % RP;
    valid = (r < RD);
    if (valid) src = imgs + ((size_t)ii * RD + r) * DD;
    dh = Ahi + (size_t)b * DD;
    dl = Alo + (size_t)b * DD;
  } else {
    const int bb = b - NIMG;
    valid = true;
    src = caps + (size_t)bb * DD;
    dh = Bhi + (size_t)bb * DD;
    dl = Blo + (size_t)bb * DD;
  }

  float x[4] = {0.f, 0.f, 0.f, 0.f};
  float ss = 0.f;
  if (valid) {
#pragma unroll
    for (int q = 0; q < 4; ++q) {
      x[q] = src[tid + q * 256];
      ss += x[q] * x[q];
    }
  }
  red[tid] = ss;
  __syncthreads();
  for (int off = 128; off > 0; off >>= 1) {
    if (tid < off) red[tid] += red[tid + off];
    __syncthreads();
  }
  const float rn = rsqrtf(red[0] + EPS_F);
#pragma unroll
  for (int q = 0; q < 4; ++q) {
    const float y = x[q] * rn;       // zero for pad rows
    const __bf16 h = (__bf16)y;
    const float  l = y - (float)h;
    dh[tid + q * 256] = h;
    dl[tid + q * 256] = (__bf16)l;
  }
}

// ---------------------------------------------------------------------------
// Kernel 2: fused per-(i,j) pipeline.
//   Phase A: S[48][48] = imgs_n[i] * caps_n[j]^T via bf16x3 WMMA (fp32-class
//            accuracy on the fast 16x16x32 bf16 pipe).
//   Phase B: 3 Euclidean-Bregman iterations; margin-sparsemax thresholds via
//            Michelot fixed point (no sort), all in LDS. Loops kept rolled to
//            keep the kernel I$-resident.
//   Phase C: sims[i][j] = sum(P * S).
// Block = 96 threads = 3 full wave32 (EXEC all-ones for WMMA).
// ---------------------------------------------------------------------------
__global__ __launch_bounds__(96) void sparseot_fused_kernel(
    const __bf16* __restrict__ Ahi, const __bf16* __restrict__ Alo,
    const __bf16* __restrict__ Bhi, const __bf16* __restrict__ Blo,
    const int* __restrict__ img_lens, const int* __restrict__ cap_lens,
    float* __restrict__ out)
{
  __shared__ float S[RP][WD + 1];   // stride 49: gcd(49,64)=1 -> conflict-free
  __shared__ float alpha_s[RP];
  __shared__ float beta_s[WD];
  __shared__ float redbuf[3];

  const int i    = blockIdx.x;
  const int j    = blockIdx.y;
  const int tid  = (int)threadIdx.x;
  const int wave = tid >> 5;        // n-tile 0..2
  const int lane = tid & 31;
  const int mrow  = lane & 15;
  const int khalf = lane >> 4;

  // ---------------- Phase A: WMMA GEMM -----------------------------------
  const __bf16* abase_h = Ahi + (size_t)i * RP * DD;
  const __bf16* abase_l = Alo + (size_t)i * RP * DD;
  const int ncol = wave * 16 + (lane & 15);
  const __bf16* bcol_h = Bhi + ((size_t)j * WD + ncol) * DD;
  const __bf16* bcol_l = Blo + ((size_t)j * WD + ncol) * DD;
  const int kb16 = khalf * 8;       // B-fragment K base within 32-slab

  v8f acc0 = {}; v8f acc1 = {}; v8f acc2 = {};

  for (int k0 = 0; k0 < DD; k0 += 32) {
    // B fragment: lane holds K = khalf*8+{0..7} and 16+khalf*8+{0..7}
    const v16bf bh = concat8(*(const bf16x8*)(bcol_h + k0 + kb16),
                             *(const bf16x8*)(bcol_h + k0 + kb16 + 16));
    const v16bf bl = concat8(*(const bf16x8*)(bcol_l + k0 + kb16),
                             *(const bf16x8*)(bcol_l + k0 + kb16 + 16));
#pragma unroll
    for (int mt = 0; mt < 3; ++mt) {
      // A fragment: lane holds K = khalf*8+{0..7} and 16+khalf*8+{0..7}
      const size_t ro = (size_t)(mt * 16 + mrow) * DD + k0 + khalf * 8;
      const v16bf ah = concat8(*(const bf16x8*)(abase_h + ro),
                               *(const bf16x8*)(abase_h + ro + 16));
      const v16bf al = concat8(*(const bf16x8*)(abase_l + ro),
                               *(const bf16x8*)(abase_l + ro + 16));
      v8f acc = (mt == 0) ? acc0 : (mt == 1) ? acc1 : acc2;
      acc = wmma_bf16(ah, bh, acc);   // hi*hi
      acc = wmma_bf16(ah, bl, acc);   // hi*lo
      acc = wmma_bf16(al, bh, acc);   // lo*hi
      if (mt == 0) acc0 = acc; else if (mt == 1) acc1 = acc; else acc2 = acc;
    }
  }

  // Spill C fragments to LDS: VGPR v of tile t -> row t*16 + khalf*8 + v
  {
    const int colS = wave * 16 + (lane & 15);
    const int rb   = khalf * 8;
#pragma unroll
    for (int v = 0; v < 8; ++v) {
      S[ 0 + rb + v][colS] = acc0[v];
      S[16 + rb + v][colS] = acc1[v];
      S[32 + rb + v][colS] = acc2[v];
    }
  }
  if (tid < RP) alpha_s[tid] = 0.f;
  if (tid < WD) beta_s[tid]  = 0.f;
  __syncthreads();

  // ---------------- Phase B: Bregman / margin-sparsemax -------------------
  const int ilen = img_lens[i];
  const int clen = cap_lens[j];
  const float mr = LAMB_F / (float)ilen;   // margin for valid image rows
  const float mc = LAMB_F / (float)clen;   // margin for valid caption cols

#pragma unroll 1
  for (int it = 0; it < 3; ++it) {
    // beta step: one thread per column w, reduce over r (n = 36)
    if (tid < WD) {
      const int w = tid;
      const float margin = (w < clen) ? mc : 0.f;
      float tau = 0.f;
      if (margin > 0.f) {              // margin==0 -> tau==0 analytically
        float sum = 0.f;
#pragma unroll 4
        for (int r = 0; r < RD; ++r) sum += alpha_s[r] + S[r][w] - 1.f;
        tau = (sum - margin) * (1.f / (float)RD);
#pragma unroll 1
        for (int pass = 0; pass < NPASS; ++pass) {   // Michelot fixed point
          float s = 0.f; int cnt = 0;
#pragma unroll 4
          for (int r = 0; r < RD; ++r) {
            const float v = alpha_s[r] + S[r][w] - 1.f;
            if (v > tau) { s += v; ++cnt; }
          }
          tau = (s - margin) / (float)(cnt > 0 ? cnt : 1);
        }
      }
      beta_s[w] = -tau;
    }
    __syncthreads();
    // alpha step: one thread per row r, reduce over w (n = 48)
    if (tid < RD) {
      const int r = tid;
      const float margin = (r < ilen) ? mr : 0.f;
      float tau = 0.f;
      if (margin > 0.f) {
        float sum = 0.f;
#pragma unroll 4
        for (int w = 0; w < WD; ++w) sum += beta_s[w] + S[r][w] - 1.f;
        tau = (sum - margin) * (1.f / (float)WD);
#pragma unroll 1
        for (int pass = 0; pass < NPASS; ++pass) {
          float s = 0.f; int cnt = 0;
#pragma unroll 4
          for (int w = 0; w < WD; ++w) {
            const float v = beta_s[w] + S[r][w] - 1.f;
            if (v > tau) { s += v; ++cnt; }
          }
          tau = (s - margin) / (float)(cnt > 0 ? cnt : 1);
        }
      }
      alpha_s[r] = -tau;
    }
    __syncthreads();
  }

  // ---------------- Phase C: sims = sum(P * fg_sims) ----------------------
  float accsim = 0.f;
#pragma unroll 1
  for (int idx = tid; idx < RD * WD; idx += 96) {
    const int r = idx / WD;
    const int w = idx - r * WD;
    const float s = S[r][w];
    float p = (beta_s[w] + alpha_s[r] - 1.f + s) * INV_LAMB;  // (b+a-M)/lamb
    p = fmaxf(p, 0.f);
    accsim += p * s;
  }
#pragma unroll
  for (int off = 16; off > 0; off >>= 1) accsim += __shfl_down(accsim, off, 32);
  if (lane == 0) redbuf[wave] = accsim;
  __syncthreads();
  if (tid == 0) out[(size_t)i * gridDim.y + j] = redbuf[0] + redbuf[1] + redbuf[2];
}

// ---------------------------------------------------------------------------
extern "C" void kernel_launch(void* const* d_in, const int* in_sizes, int n_in,
                              void* d_out, int out_size, void* d_ws, size_t ws_size,
                              hipStream_t stream) {
  (void)in_sizes; (void)n_in; (void)out_size; (void)ws_size;
  // setup_inputs order: img_cls, imgs, cap_cls, caps, img_lens, cap_lens
  const float* imgs     = (const float*)d_in[1];
  const float* caps     = (const float*)d_in[3];
  const int*   img_lens = (const int*)d_in[4];
  const int*   cap_lens = (const int*)d_in[5];
  float*       out      = (float*)d_out;

  // Workspace: 4 bf16 arrays of 128*48*1024 elements (~50 MB total)
  const size_t elems = (size_t)BI * RP * DD;
  __bf16* Ahi = (__bf16*)d_ws;
  __bf16* Alo = Ahi + elems;
  __bf16* Bhi = Alo + elems;
  __bf16* Blo = Bhi + elems;

  normsplit_kernel<<<dim3(BI * RP + BT * WD), dim3(256), 0, stream>>>(
      imgs, caps, Ahi, Alo, Bhi, Blo);
  sparseot_fused_kernel<<<dim3(BI, BT), dim3(96), 0, stream>>>(
      Ahi, Alo, Bhi, Blo, img_lens, cap_lens, out);
}
